// ContextualLoss_73306501808909
// MI455X (gfx1250) — matrix operand order repaired and checked by reading, not compiled, not run on hardware
//
#include <hip/hip_runtime.h>

// Problem constants (fixed by the reference): b=64, c=256, h=w=32 -> n=1024.
#define BATCH   64
#define C_DIM   256
#define N_SAMP  1024
#define EPSV    1e-5f

typedef __bf16 bf16_t;
typedef bf16_t v8bf  __attribute__((ext_vector_type(8)));
typedef bf16_t v16bf __attribute__((ext_vector_type(16)));
typedef float  v8f   __attribute__((ext_vector_type(8)));

// Workspace layout (bytes). Total ~68 MB.
#define WS_YT   ((size_t)0)                      // 64*1024*256 bf16 = 32 MB
#define WS_XT   ((size_t)33554432)               // 32 MB
#define WS_INVY ((size_t)67108864)               // 64*1024 f32 = 256 KB
#define WS_INVX (WS_INVY + 262144)
#define WS_CXI  (WS_INVX + 262144)               // 64*1024 f32
#define WS_BL   (WS_CXI + 262144)                // 64 f32

__device__ __forceinline__ unsigned short f2bf(float f) {
  unsigned u = __float_as_uint(f);
  u += 0x7FFFu + ((u >> 16) & 1u);   // round-to-nearest-even
  return (unsigned short)(u >> 16);
}

// ---------------------------------------------------------------------------
// 1) inv_norm[b][n] = 1 / max(||v[b,:,n]||_2, eps), for y and x.
//    grid (256, 2), block 256. Loads are fully coalesced across n.
// ---------------------------------------------------------------------------
__global__ __launch_bounds__(256) void norm_kernel(
    const float* __restrict__ y, const float* __restrict__ x,
    float* __restrict__ invy, float* __restrict__ invx) {
  const float* src = blockIdx.y ? x : y;
  float*       dst = blockIdx.y ? invx : invy;
  const int b = blockIdx.x >> 2;
  const int n = ((blockIdx.x & 3) << 8) + threadIdx.x;
  const float* p = src + (size_t)b * C_DIM * N_SAMP + n;
  float ss = 0.0f;
#pragma unroll 8
  for (int c = 0; c < C_DIM; ++c) {
    float v = p[(size_t)c * N_SAMP];
    ss += v * v;
  }
  dst[(size_t)b * N_SAMP + n] = 1.0f / fmaxf(sqrtf(ss), EPSV);
}

// ---------------------------------------------------------------------------
// 2) Transpose+scale+convert: Yt/Xt[b][n][c] = src[b][c][n] * inv_norm (bf16).
//    32x32 LDS tile, conflict-free (stride 33). grid (64*32*8, 2), block 256.
// ---------------------------------------------------------------------------
__global__ __launch_bounds__(256) void transpose_kernel(
    const float* __restrict__ y, const float* __restrict__ x,
    const float* __restrict__ invy, const float* __restrict__ invx,
    unsigned short* __restrict__ Yt, unsigned short* __restrict__ Xt) {
  __shared__ float tile[32][33];
  const float*    src  = blockIdx.y ? x : y;
  const float*    invn = blockIdx.y ? invx : invy;
  unsigned short* dst  = blockIdx.y ? Xt : Yt;

  const int id = blockIdx.x;
  const int ct = id & 7;            // 8 c-tiles of 32
  const int nt = (id >> 3) & 31;    // 32 n-tiles of 32
  const int b  = id >> 8;
  const int n0 = nt * 32, c0 = ct * 32;
  const int tid = threadIdx.x;

  const int j  = tid & 31;          // n within tile (coalesced read)
  const int i0 = tid >> 5;          // 0..7
  const float inv = invn[(size_t)b * N_SAMP + n0 + j];
#pragma unroll
  for (int p = 0; p < 4; ++p) {
    const int ci = i0 + p * 8;
    tile[ci][j] = src[((size_t)b * C_DIM + c0 + ci) * N_SAMP + n0 + j] * inv;
  }
  __syncthreads();
  const int cc = tid & 31;          // c within tile (coalesced write)
  const int r0 = tid >> 5;
#pragma unroll
  for (int p = 0; p < 4; ++p) {
    const int nn = r0 + p * 8;
    dst[((size_t)b * N_SAMP + n0 + nn) * C_DIM + c0 + cc] = f2bf(tile[cc][nn]);
  }
}

// ---------------------------------------------------------------------------
// 3) Per (batch, 16-row strip): WMMA GEMM 16x1024x256 into a 64 KB LDS strip,
//    then row-wise min / sum-exp reductions -> cx_i[b][row].
//    grid (64, 64), block 256 (8 waves; wave w owns columns [w*128, w*128+128)).
// ---------------------------------------------------------------------------
__global__ __launch_bounds__(256) void gemm_cx_kernel(
    const bf16_t* __restrict__ Yt, const bf16_t* __restrict__ Xt,
    float* __restrict__ cxi) {
  __shared__ float sdist[16 * N_SAMP];   // 64 KB dist strip

  const int tid  = threadIdx.x;
  const int lane = tid & 31;
  const int wave = tid >> 5;
  const int b    = blockIdx.x;
  const int m0   = blockIdx.y * 16;
  const int half = lane >> 4;            // lane group (0/1)
  const int l16  = lane & 15;

  // A strip resident in VGPRs: row M = lane%16; per lane two contiguous 16 B
  // chunks per 32-K step (documented 16-bit A 16x32 layout).
  const bf16_t* Arow = Yt + ((size_t)b * N_SAMP + m0 + l16) * C_DIM;
  v16bf a[8];
#pragma unroll
  for (int ks = 0; ks < 8; ++ks) {
    const int k0 = ks * 32;
    v8bf lo = *(const v8bf*)(Arow + k0 + half * 8);
    v8bf hi = *(const v8bf*)(Arow + k0 + 16 + half * 8);
    a[ks] = __builtin_shufflevector(lo, hi, 0,1,2,3,4,5,6,7,8,9,10,11,12,13,14,15);
  }

#pragma unroll 1
  for (int t = 0; t < 8; ++t) {
    const int n0 = wave * 128 + t * 16;
    // B 32x16: column N = lane%16; lane group g holds K = 32*ks + g*8 .. +7 and 32*ks + 16 + g*8 .. +7.
    const bf16_t* Brow = Xt + ((size_t)b * N_SAMP + n0 + l16) * C_DIM;
    v8f acc = {0.f, 0.f, 0.f, 0.f, 0.f, 0.f, 0.f, 0.f};
#pragma unroll
    for (int ks = 0; ks < 8; ++ks) {
      const int k0 = ks * 32 + half * 8;
      v8bf lo = *(const v8bf*)(Brow + k0);
      v8bf hi = *(const v8bf*)(Brow + k0 + 16);
      v16bf bb = __builtin_shufflevector(lo, hi, 0,1,2,3,4,5,6,7,8,9,10,11,12,13,14,15);
      acc = __builtin_amdgcn_wmma_f32_16x16x32_bf16(
          /*neg_a=*/false, a[ks], /*neg_b=*/false, bb,
          /*c_mod=*/(short)0, acc, /*reuse_a=*/false, /*reuse_b=*/false);
    }
    // C/D layout: lane L holds column N=L%16; VGPR v -> row M = v + (L/16)*8.
    const int col = n0 + l16;
    const int rb  = half * 8;
#pragma unroll
    for (int v = 0; v < 8; ++v) {
      float d = 1.0f - acc[v];
      d = fminf(fmaxf(d, 0.0f), 2.0f);
      sdist[(rb + v) * N_SAMP + col] = d;
    }
  }
  __syncthreads();

  // Row reductions: 16 threads per row (16 rows x 16 threads = 256).
  const int row = tid >> 4;
  const int cb  = tid & 15;
  const float* rp = &sdist[row * N_SAMP];

  float mn = 3.0e38f;
  for (int c2 = cb; c2 < N_SAMP; c2 += 16) mn = fminf(mn, rp[c2]);
#pragma unroll
  for (int off = 8; off; off >>= 1) mn = fminf(mn, __shfl_xor(mn, off, 16));

  const float dmin  = mn;
  const float dminc = fmaxf(dmin, EPSV);
  const float invd  = 1.0f / dminc;

  float s = 0.0f;
  for (int c2 = cb; c2 < N_SAMP; c2 += 16) {
    float lw = (1.0f - rp[c2] * invd) * 10.0f;   // /BANDWIDTH(0.1)
    lw = fminf(fmaxf(lw, -50.0f), 50.0f);
    s += __expf(lw);
  }
#pragma unroll
  for (int off = 8; off; off >>= 1) s += __shfl_xor(s, off, 16);

  if (cb == 0) {
    // max_j cx = exp(clip((1 - dmin/dminc)/0.1)) / (sum + eps)
    float lwm = fminf(fmaxf((1.0f - dmin * invd) * 10.0f, -50.0f), 50.0f);
    cxi[(size_t)b * N_SAMP + m0 + row] = __expf(lwm) / (s + EPSV);
  }
}

// ---------------------------------------------------------------------------
// 4) Per-batch mean + -log. grid 64, block 256. Deterministic tree reduce.
// ---------------------------------------------------------------------------
__global__ __launch_bounds__(256) void batch_mean_kernel(
    const float* __restrict__ cxi, float* __restrict__ bl) {
  __shared__ float red[8];
  const int b = blockIdx.x, tid = threadIdx.x;
  float s = 0.0f;
  for (int i = tid; i < N_SAMP; i += 256) s += cxi[(size_t)b * N_SAMP + i];
#pragma unroll
  for (int off = 16; off; off >>= 1) s += __shfl_xor(s, off, 32);
  if ((tid & 31) == 0) red[tid >> 5] = s;
  __syncthreads();
  if (tid == 0) {
    float t = 0.0f;
#pragma unroll
    for (int w = 0; w < 8; ++w) t += red[w];
    bl[b] = -__logf(t * (1.0f / N_SAMP) + EPSV);
  }
}

// ---------------------------------------------------------------------------
// 5) Final mean over 64 batches -> scalar loss. One wave32.
// ---------------------------------------------------------------------------
__global__ __launch_bounds__(32) void final_kernel(
    const float* __restrict__ bl, float* __restrict__ out) {
  const int lane = threadIdx.x;
  float s = bl[lane] + bl[lane + 32];
#pragma unroll
  for (int off = 16; off; off >>= 1) s += __shfl_xor(s, off, 32);
  if (lane == 0) out[0] = s * (1.0f / BATCH);
}

// ---------------------------------------------------------------------------
extern "C" void kernel_launch(void* const* d_in, const int* in_sizes, int n_in,
                              void* d_out, int out_size, void* d_ws, size_t ws_size,
                              hipStream_t stream) {
  const float* y = (const float*)d_in[0];
  const float* x = (const float*)d_in[1];
  float* out = (float*)d_out;
  char*  ws  = (char*)d_ws;   // needs ~68 MB

  unsigned short* YtU = (unsigned short*)(ws + WS_YT);
  unsigned short* XtU = (unsigned short*)(ws + WS_XT);
  const bf16_t*   Yt  = (const bf16_t*)(ws + WS_YT);
  const bf16_t*   Xt  = (const bf16_t*)(ws + WS_XT);
  float* invy = (float*)(ws + WS_INVY);
  float* invx = (float*)(ws + WS_INVX);
  float* cxi  = (float*)(ws + WS_CXI);
  float* bl   = (float*)(ws + WS_BL);

  norm_kernel<<<dim3(256, 2), 256, 0, stream>>>(y, x, invy, invx);
  transpose_kernel<<<dim3(BATCH * 32 * 8, 2), 256, 0, stream>>>(y, x, invy, invx, YtU, XtU);
  gemm_cx_kernel<<<dim3(BATCH, N_SAMP / 16), 256, 0, stream>>>(Yt, Xt, cxi);
  batch_mean_kernel<<<BATCH, 256, 0, stream>>>(cxi, bl);
  final_kernel<<<1, 32, 0, stream>>>(bl, out);
}
